// SparseAutoEncoder_84078279787084
// MI455X (gfx1250) — compile-verified, matrix operand order checked by reading, not run on hardware
//
#include <hip/hip_runtime.h>
#include <hip/hip_bf16.h>

typedef __attribute__((ext_vector_type(16))) _Float16 v16h;
typedef __attribute__((ext_vector_type(8)))  float    v8f;
typedef __attribute__((ext_vector_type(4)))  float    v4f;

#define B_SZ 64   // batch
#define HW   64   // hyper hidden width (K of the WMMA GEMM)

// Build one 16x32-f16 fragment (8 VGPRs) from a row of 64 f32 values.
// Per ISA 16-bit A/B layout: lane-half "half" selects K base 0 or 8; elements
// 0..7 -> K = basej..basej+7, elements 8..15 -> K = basej+16..basej+23.
__device__ __forceinline__ v16h load_frag_f32_to_f16(const float* __restrict__ row, int basej) {
  v4f f0 = *(const v4f*)(row + basej + 0);
  v4f f1 = *(const v4f*)(row + basej + 4);
  v4f f2 = *(const v4f*)(row + basej + 16);
  v4f f3 = *(const v4f*)(row + basej + 20);
  v16h o;
#pragma unroll
  for (int i = 0; i < 4; ++i) {
    o[i]      = (_Float16)f0[i];
    o[i + 4]  = (_Float16)f1[i];
    o[i + 8]  = (_Float16)f2[i];
    o[i + 12] = (_Float16)f3[i];
  }
  return o;
}

// Tiny hyper MLP: h = tanh(tanh(mu @ W0.T + b0) @ W1.T + b1).  64 threads, 1 block.
__global__ void __launch_bounds__(64)
hyper_kernel(const float* __restrict__ mu, const float* __restrict__ W0,
             const float* __restrict__ b0, const float* __restrict__ W1,
             const float* __restrict__ b1, float* __restrict__ h_out) {
  __shared__ float h0[B_SZ * HW];
  const int b = threadIdx.x;
  const float m0 = mu[b * 4 + 0], m1 = mu[b * 4 + 1];
  const float m2 = mu[b * 4 + 2], m3 = mu[b * 4 + 3];
  for (int i = 0; i < HW; ++i) {
    float s = b0[i] + m0 * W0[i * 4 + 0] + m1 * W0[i * 4 + 1]
                    + m2 * W0[i * 4 + 2] + m3 * W0[i * 4 + 3];
    h0[b * HW + i] = tanhf(s);
  }
  __syncthreads();
  for (int i = 0; i < HW; ++i) {
    float s = b1[i];
#pragma unroll 8
    for (int j = 0; j < HW; ++j) s += h0[b * HW + j] * W1[i * HW + j];
    h_out[b * HW + i] = tanhf(s);
  }
}

// One-off transpose: in (64 x nk, row-major) -> outT (nk x 64, k-major).
__global__ void __launch_bounds__(256)
transpose_kernel(const float* __restrict__ in, float* __restrict__ outT, int nk) {
  const int idx = blockIdx.x * 256 + threadIdx.x;   // idx = k*64 + b
  if (idx < nk * B_SZ) {
    const int k = idx >> 6;
    const int b = idx & 63;
    outT[idx] = in[b * nk + k];
  }
}

// One 16-row W2 tile: P = h @ W2tile.T via 2 accumulating WMMAs per M-tile,
// then multiply by xT[k, b] and reduce into acc.  MASKED=false compiles away
// all bounds handling for full tiles (the dominant case).
template <bool MASKED>
__device__ __forceinline__ void
do_tile(int kbase, int ni, int lane, int half, int boff,
        const float* __restrict__ Wc, const float* __restrict__ xT,
        const v16h (&afrag)[4][2], float (&acc)[4][8], bool prefetch_next)
{
  int r = kbase + (lane & 15);
  if (MASKED && r > ni - 1) r = ni - 1;     // clamp OOB rows (x masked to 0)
  const float* wrow = Wc + (long long)r * HW;

  if (prefetch_next)
    __builtin_prefetch(wrow + 4LL * 16 * HW, 0, 1);

  // B fragments: lanes = W2 rows (N), elements = hidden j (K)
  v16h bf0 = load_frag_f32_to_f16(wrow, half);
  v16h bf1 = load_frag_f32_to_f16(wrow, 32 + half);

  const int k  = kbase + (lane & 15);
  const int kc = MASKED ? ((k < ni) ? k : 0) : k;
  const float in_range = (!MASKED || k < ni) ? 1.f : 0.f;
  const float* xlane = xT + (long long)kc * B_SZ + boff;

#pragma unroll
  for (int m = 0; m < 4; ++m) {
    v8f d = {};
    d = __builtin_amdgcn_wmma_f32_16x16x32_f16(false, afrag[m][0], false, bf0,
                                               (short)0, d, false, false);
    d = __builtin_amdgcn_wmma_f32_16x16x32_f16(false, afrag[m][1], false, bf1,
                                               (short)0, d, false, false);
    v4f xq0 = *(const v4f*)(xlane + m * 16);
    v4f xq1 = *(const v4f*)(xlane + m * 16 + 4);
    if (MASKED) { xq0 = xq0 * in_range; xq1 = xq1 * in_range; }
    // D element (v, lane): b = m*16 + boff + v, k = kbase + (lane&15)
#pragma unroll
    for (int v = 0; v < 4; ++v) {
      acc[m][v]     += d[v]     * xq0[v];
      acc[m][v + 4] += d[v + 4] * xq1[v];
    }
  }
}

// Fused hyper-head + per-sample FC layer (x in k-major layout):
//   out[b,c] = act( sum_k xT[k,b] * (h[b,:]·W2[cut + c*ni + k,:])
//                   + h[b,:]·W2[cut + ni*no + c,:] + b2[cut + ni*no + c] )
// One workgroup per output channel c; 4 waves stride 16-row W2 tiles.
__global__ void __launch_bounds__(128)
fused_hyper_fc(const float* __restrict__ xT,   // ni x B (k-major)
               const float* __restrict__ W2,   // R x 64
               const float* __restrict__ b2,   // R
               const float* __restrict__ h,    // B x 64
               float* __restrict__ outT,       // no x B (k-major, next layer) or null
               float* __restrict__ out_row,    // B x no (row-major final) or null
               int ni, int no, long long cut, int act_tanh)
{
  __shared__ float s_bias[B_SZ];
  __shared__ float s_part[4 * B_SZ];

  const int c    = blockIdx.x;
  const int tid  = threadIdx.x;
  const int wave = tid >> 5;
  const int lane = tid & 31;
  const int half = (lane & 16) ? 8 : 0;   // K-base select for this lane half
  const int boff = (lane & 16) ? 8 : 0;   // M offset in D fragments

  // ---- bias row (scalar): h[b,:] . W2[cut + ni*no + c, :] + b2 ----
  if (tid < B_SZ) {
    const long long br = cut + (long long)ni * no + c;
    const float* wrow = W2 + br * HW;
    float s = b2[br];
#pragma unroll 8
    for (int j = 0; j < HW; ++j) s += h[tid * HW + j] * wrow[j];
    s_bias[tid] = s;
  }

  // ---- A fragments (h, reused for every tile): 4 M-tiles x 2 K-steps ----
  v16h afrag[4][2];
#pragma unroll
  for (int m = 0; m < 4; ++m) {
    const float* hrow = h + (m * 16 + (lane & 15)) * HW;
#pragma unroll
    for (int s = 0; s < 2; ++s)
      afrag[m][s] = load_frag_f32_to_f16(hrow, s * 32 + half);
  }

  float acc[4][8];
#pragma unroll
  for (int m = 0; m < 4; ++m)
#pragma unroll
    for (int v = 0; v < 8; ++v) acc[m][v] = 0.f;

  const float* Wc = W2 + (cut + (long long)c * ni) * HW;
  const int nfull = ni >> 4;                 // full 16-row tiles (no masking)

  for (int t = wave; t < nfull; t += 4)
    do_tile<false>(t * 16, ni, lane, half, boff, Wc, xT, afrag, acc,
                   t + 4 < nfull);

  if ((ni & 15) && wave == (nfull & 3))      // single masked tail tile
    do_tile<true>(nfull * 16, ni, lane, half, boff, Wc, xT, afrag, acc, false);

  // reduce over k within each 16-lane half (butterfly)
#pragma unroll
  for (int m = 0; m < 4; ++m)
#pragma unroll
    for (int v = 0; v < 8; ++v) {
#pragma unroll
      for (int sft = 8; sft >= 1; sft >>= 1)
        acc[m][v] += __shfl_xor(acc[m][v], sft, 32);
    }

  if ((lane & 15) == 0) {
#pragma unroll
    for (int m = 0; m < 4; ++m)
#pragma unroll
      for (int v = 0; v < 8; ++v)
        s_part[wave * B_SZ + m * 16 + boff + v] = acc[m][v];
  }
  __syncthreads();

  if (tid < B_SZ) {
    float tot = s_bias[tid];
#pragma unroll
    for (int w = 0; w < 4; ++w) tot += s_part[w * B_SZ + tid];
    if (act_tanh) tot = tanhf(tot);
    if (outT)    outT[(long long)c * B_SZ + tid] = tot;      // coalesced
    if (out_row) out_row[(long long)tid * no + c] = tot;
  }
}

extern "C" void kernel_launch(void* const* d_in, const int* in_sizes, int n_in,
                              void* d_out, int out_size, void* d_ws, size_t ws_size,
                              hipStream_t stream) {
  (void)in_sizes; (void)n_in; (void)out_size; (void)ws_size;

  const float* z    = (const float*)d_in[0];
  const float* mu   = (const float*)d_in[1];
  const float* enW0 = (const float*)d_in[2];
  const float* enb0 = (const float*)d_in[3];
  const float* enW1 = (const float*)d_in[4];
  const float* enb1 = (const float*)d_in[5];
  const float* enW2 = (const float*)d_in[6];
  const float* enb2 = (const float*)d_in[7];
  const float* deW0 = (const float*)d_in[8];
  const float* deb0 = (const float*)d_in[9];
  const float* deW1 = (const float*)d_in[10];
  const float* deb1 = (const float*)d_in[11];
  const float* deW2 = (const float*)d_in[12];
  const float* deb2 = (const float*)d_in[13];

  float* ws   = (float*)d_ws;
  float* h_en = ws;               // 64*64      = 4096
  float* h_de = ws + 4096;        // 64*64      = 4096
  float* zT   = ws + 8192;        // 2000*64    = 128000
  float* x1T  = ws + 136192;      // 200*64     = 12800
  float* x2T  = ws + 148992;      // 100*64     = 6400
  float* latT = ws + 155392;      // 10*64      = 640
  float* y1T  = ws + 156032;      // 100*64     = 6400
  float* y2T  = ws + 162432;      // 200*64     = 12800

  float* z_rec = (float*)d_out;           // 64*2000 row-major
  float* x_out = z_rec + 64 * 2000;       // 64*10   row-major

  hyper_kernel<<<1, 64, 0, stream>>>(mu, enW0, enb0, enW1, enb1, h_en);
  hyper_kernel<<<1, 64, 0, stream>>>(mu, deW0, deb0, deW1, deb1, h_de);
  transpose_kernel<<<(2000 * B_SZ + 255) / 256, 256, 0, stream>>>(z, zT, 2000);

  // encoder: (2000->200 linear), (200->100 tanh), (100->10 linear)
  fused_hyper_fc<<<200,  128, 0, stream>>>(zT,  enW2, enb2, h_en, x1T,  nullptr, 2000, 200,  0LL,      0);
  fused_hyper_fc<<<100,  128, 0, stream>>>(x1T, enW2, enb2, h_en, x2T,  nullptr, 200,  100,  400200LL, 1);
  fused_hyper_fc<<<10,   128, 0, stream>>>(x2T, enW2, enb2, h_en, latT, x_out,   100,  10,   420300LL, 0);
  // decoder: (10->100 linear), (100->200 tanh), (200->2000 linear)
  fused_hyper_fc<<<100,  128, 0, stream>>>(latT, deW2, deb2, h_de, y1T, nullptr, 10,   100,  0LL,      0);
  fused_hyper_fc<<<200,  128, 0, stream>>>(y1T,  deW2, deb2, h_de, y2T, nullptr, 100,  200,  1100LL,   1);
  fused_hyper_fc<<<2000, 128, 0, stream>>>(y2T,  deW2, deb2, h_de, nullptr, z_rec, 200, 2000, 21300LL, 0);
}
